// CharRNN_68212670595566
// MI455X (gfx1250) — matrix operand order repaired and checked by reading, not compiled, not run on hardware
//
#include <hip/hip_runtime.h>
#include <hip/hip_bf16.h>

// ---------------------------------------------------------------------------
// CharRNN (4-layer residual LSTM) for MI455X / gfx1250, wave32 + WMMA bf16.
// V=96 D=24 H=200 L=4 B=64 T=512.  Padded: Hp=208 (13 n-tiles), 4Hp=832
// (52 n-tiles), K: D->32 (2 ksteps), H->224 (7 ksteps).  M = T*B = 32768.
// ---------------------------------------------------------------------------

typedef __bf16 bf16;
typedef unsigned int u32;
typedef __bf16 v16bf __attribute__((ext_vector_type(16)));
typedef float  v8f   __attribute__((ext_vector_type(8)));
typedef u32    u32x4 __attribute__((ext_vector_type(4)));

union BFrag { v16bf v; u32x4 q[2]; };

__device__ __forceinline__ bf16 f2bf(float x) {
    union { float f; u32 u; } a; a.f = x;
    u32 r = a.u + 0x7fffu + ((a.u >> 16) & 1u);           // round-nearest-even
    unsigned short s = (unsigned short)(r >> 16);
    union { unsigned short s; bf16 b; } o; o.s = s;
    return o.b;
}

// fast sigmoid / tanh on hardware transcendentals (v_exp_f32 + v_rcp_f32)
__device__ __forceinline__ float sigm(float x) {
    return __builtin_amdgcn_rcpf(1.0f + __expf(-x));
}
__device__ __forceinline__ float tanh_fast(float x) {
    return 2.0f * sigm(2.0f * x) - 1.0f;
}

__device__ __forceinline__ v8f wmma_bf(v16bf a, v16bf b, v8f c) {
    // emits v_wmma_f32_16x16x32_bf16
    return __builtin_amdgcn_wmma_f32_16x16x32_bf16(false, a, false, b, (short)0, c,
                                                   false, false);
}

// A fragment (16x32 bf16) from row-major matrix (global or LDS), per ISA layout:
// lane m = l&15 holds row m; elems 0..7 -> K = kbase + hi*8 + j,
// elems 8..15 -> K = kbase + 16 + hi*8 + (j-8).
__device__ __forceinline__ v16bf loadA(const bf16* A, int row0, int lda, int kbase,
                                       int lane) {
    int m = lane & 15, hi = lane >> 4;
    const bf16* p = A + (size_t)(row0 + m) * lda + kbase + hi * 8;
    BFrag f;
    f.q[0] = *(const u32x4*)p;
    f.q[1] = *(const u32x4*)(p + 16);
    return f.v;
}

// B fragment from pre-swizzled buffer: fragment (ks,nt) is a 1KB block, each
// lane's 16 bf16 contiguous (two 16B loads).
__device__ __forceinline__ v16bf loadB(const bf16* Bsw, int ks, int nt, int NT,
                                       int lane) {
    const bf16* p = Bsw + (((size_t)(ks * NT + nt) * 32 + lane) << 4);
    BFrag f;
    f.q[0] = *(const u32x4*)p;
    f.q[1] = *(const u32x4*)(p + 8);
    return f.v;
}

// ---------------------------------------------------------------------------
// Prep kernels
// ---------------------------------------------------------------------------

// embT gather: emb_bf16[(t*64+b)*32 + k] = k<24 ? C[xb[b,t],k] : 0
__global__ void k_embed(const int* __restrict__ xb, const float* __restrict__ C,
                        bf16* __restrict__ emb) {
    int i = blockIdx.x * 256 + threadIdx.x;     // < 32768*32
    int row = i >> 5, k = i & 31;
    int t = row >> 6, b = row & 63;
    float v = 0.f;
    if (k < 24) v = C[(size_t)xb[b * 512 + t] * 24 + k];
    emb[i] = f2bf(v);
}

// Swizzle fp32 [srcK, srcN] row-major into WMMA-B fragment layout with K pad to
// 32*KS and N pad to 16*NT.  gatePerm: output col N -> gate q=N/208, j=N%208,
// source col q*200+j (zero for j>=200).
__global__ void k_swz(const float* __restrict__ src, bf16* __restrict__ dst,
                      int srcK, int srcN, int NT, int gatePerm) {
    int i = blockIdx.x * 256 + threadIdx.x;     // < KS*NT*512
    int j = i & 15, lane = (i >> 4) & 31, frag = i >> 9;
    int ks = frag / NT, nt = frag % NT;
    int K = ks * 32 + (j >> 3) * 16 + (lane >> 4) * 8 + (j & 7);
    int N = nt * 16 + (lane & 15);
    float v = 0.f;
    if (gatePerm) {
        int q = N / 208, jj = N - 208 * q;
        if (K < srcK && jj < 200) v = src[(size_t)K * srcN + q * 200 + jj];
    } else {
        if (K < srcK && N < srcN) v = src[(size_t)K * srcN + N];
    }
    dst[i] = f2bf(v);
}

// Gate-permuted biases: bpre[4][832] (layer0=U_in_b, 1..3=Uh_b), bex[3][208].
__global__ void k_bias(const float* __restrict__ uinb, const float* __restrict__ uhb,
                       const float* __restrict__ vhb, float* __restrict__ bpre,
                       float* __restrict__ bex) {
    int i = blockIdx.x * 256 + threadIdx.x;
    if (i < 4 * 832) {
        int layer = i / 832, col = i % 832, q = col / 208, j = col % 208;
        const float* src = (layer == 0) ? uinb : (uhb + (layer - 1) * 800);
        bpre[i] = (j < 200) ? src[q * 200 + j] : 0.f;
    } else if (i < 4 * 832 + 3 * 208) {
        int k = i - 4 * 832, n = k / 208, j = k % 208;
        bex[k] = (j < 200) ? vhb[n * 200 + j] : 0.f;
    }
}

// ---------------------------------------------------------------------------
// Generic GEMM: out[32768, 16*NT] = act(A[32768, 32*KS]bf16 @ Bsw + bias)
// One wave per 16x16 tile; block = 8 waves; grid sized exactly.
// ---------------------------------------------------------------------------
__global__ void k_gemm(const bf16* __restrict__ A, int lda,
                       const bf16* __restrict__ Bsw, int NT, int KS,
                       const float* __restrict__ bias, float* __restrict__ out,
                       int ldo, int act) {
    int lane = threadIdx.x & 31;
    int gwave = blockIdx.x * 8 + (threadIdx.x >> 5);
    int mt = gwave / NT, nt = gwave % NT;
    v8f acc = {0.f, 0.f, 0.f, 0.f, 0.f, 0.f, 0.f, 0.f};
    for (int ks = 0; ks < KS; ks++) {
        v16bf a = loadA(A, mt * 16, lda, ks * 32, lane);
        v16bf b = loadB(Bsw, ks, nt, NT, lane);
        acc = wmma_bf(a, b, acc);
    }
    int col = nt * 16 + (lane & 15), hi = lane >> 4;
    float bv = bias[col];
#pragma unroll
    for (int r = 0; r < 8; r++) {
        int row = mt * 16 + hi * 8 + r;
        float v = acc[r] + bv;
        if (act) v = tanh_fast(v);
        out[(size_t)row * ldo + col] = v;
    }
}

// ---------------------------------------------------------------------------
// Persistent single-WGP LSTM layer.  832 threads = 26 waves.  Wave w owns
// h-tile ht = w%13 and M-tiles {2p, 2p+1} (p = w/13): the two M-tiles share
// every weight fragment, so each B fragment load feeds two WMMAs (28 B-frag
// loads/wave/step instead of 56).  c lives in registers across all 512 steps;
// h_{t-1} sits in a 28KB bf16 LDS tile read as WMMA A-fragments (ds_load_b128).
// ---------------------------------------------------------------------------
__global__ void __launch_bounds__(832)
k_lstm(const float* __restrict__ pre, const bf16* __restrict__ Wsw,
       const float* __restrict__ ex, const float* __restrict__ h0,
       const float* __restrict__ c0, bf16* __restrict__ hout) {
    __shared__ bf16 hl[64 * 224];
    const int tid = threadIdx.x, lane = tid & 31, wave = tid >> 5;
    const int hi = lane >> 4, c16 = lane & 15;

    // init h_{-1} = h0 (broadcast over batch), K-pad cols zero
    for (int i = tid; i < 64 * 224; i += 832) {
        int col = i % 224;
        hl[i] = (col < 200) ? f2bf(h0[col]) : f2bf(0.f);
    }

    const int mtp = wave / 13;          // 0..1 -> M-tiles {2p, 2p+1}
    const int ht  = wave % 13;          // h-tile (shared weight columns)
    const int hcol = ht * 16 + c16;

    float c[2][8];
    {
        float cv = (hcol < 200) ? c0[hcol] : 0.f;
#pragma unroll
        for (int i = 0; i < 2; i++)
#pragma unroll
            for (int r = 0; r < 8; r++) c[i][r] = cv;
    }
    __syncthreads();

    const v8f vz = {0.f, 0.f, 0.f, 0.f, 0.f, 0.f, 0.f, 0.f};
    for (int t = 0; t < 512; t++) {
        // pull this step's 208KB gate-preactivation slab toward the WGP while
        // the WMMA phase runs: 832 lanes x 256B apart covers the whole slab.
        __builtin_prefetch(pre + (size_t)t * 64 * 832 + (size_t)tid * 64, 0, 1);

        v8f acc[2][4];
#pragma unroll
        for (int i = 0; i < 2; i++)
#pragma unroll
            for (int q = 0; q < 4; q++) acc[i][q] = vz;

        for (int ks = 0; ks < 7; ks++) {
            v16bf a0 = loadA(hl, (mtp * 2 + 0) * 16, 224, ks * 32, lane);
            v16bf a1 = loadA(hl, (mtp * 2 + 1) * 16, 224, ks * 32, lane);
#pragma unroll
            for (int q = 0; q < 4; q++) {
                v16bf b = loadB(Wsw, ks, q * 13 + ht, 52, lane);
                acc[0][q] = wmma_bf(a0, b, acc[0][q]);
                acc[1][q] = wmma_bf(a1, b, acc[1][q]);
            }
        }
        __syncthreads();   // all reads of h_{t-1} done

        for (int i = 0; i < 2; i++) {
            int mt = mtp * 2 + i;
#pragma unroll
            for (int r = 0; r < 8; r++) {
                int row = mt * 16 + hi * 8 + r;
                size_t prow = ((size_t)t * 64 + row) * 832;
                float gi = acc[i][0][r] + pre[prow + 0 * 208 + hcol];
                float gf = acc[i][1][r] + pre[prow + 1 * 208 + hcol];
                float gg = acc[i][2][r] + pre[prow + 2 * 208 + hcol];
                float go = acc[i][3][r] + pre[prow + 3 * 208 + hcol];
                float cn = sigm(gf) * c[i][r] + sigm(gi) * tanh_fast(gg);
                c[i][r] = cn;
                float ev = ex ? ex[((size_t)t * 64 + row) * 208 + hcol] : 0.f;
                float hv = sigm(go) * tanh_fast(cn) + ev;
                if (hcol >= 200) hv = 0.f;
                bf16 hb = f2bf(hv);
                hl[row * 224 + hcol] = hb;
                hout[((size_t)t * 64 + row) * 224 + hcol] = hb;
            }
        }
        __syncthreads();   // h_t visible before next step's fragment loads
    }
}

// ---------------------------------------------------------------------------
// logits[b,t,v] = sum_l h_all[l]@Why[l] + b_y.  K = 4 layers x 7 ksteps.
// ---------------------------------------------------------------------------
__global__ void k_logits(const bf16* __restrict__ hall, const bf16* __restrict__ Ysw,
                         const float* __restrict__ by, float* __restrict__ out) {
    int lane = threadIdx.x & 31;
    int gwave = blockIdx.x * 8 + (threadIdx.x >> 5);
    int mt = gwave / 6, nt = gwave % 6;
    v8f acc = {0.f, 0.f, 0.f, 0.f, 0.f, 0.f, 0.f, 0.f};
    for (int l = 0; l < 4; l++) {
        const bf16* A  = hall + (size_t)l * 32768 * 224;
        const bf16* Bl = Ysw + (size_t)l * 7 * 6 * 512;
        for (int ks = 0; ks < 7; ks++) {
            v16bf a = loadA(A, mt * 16, 224, ks * 32, lane);
            v16bf b = loadB(Bl, ks, nt, 6, lane);
            acc = wmma_bf(a, b, acc);
        }
    }
    int col = nt * 16 + (lane & 15), hi = lane >> 4;
    float bv = by[col];
#pragma unroll
    for (int r = 0; r < 8; r++) {
        int row = mt * 16 + hi * 8 + r;
        int tt = row >> 6, bb = row & 63;
        out[(size_t)bb * 512 * 96 + (size_t)tt * 96 + col] = acc[r] + bv;
    }
}

// ---------------------------------------------------------------------------
// Host launcher
// ---------------------------------------------------------------------------
extern "C" void kernel_launch(void* const* d_in, const int* in_sizes, int n_in,
                              void* d_out, int out_size, void* d_ws, size_t ws_size,
                              hipStream_t stream) {
    (void)in_sizes; (void)n_in; (void)out_size; (void)ws_size;
    const int M = 32768;                 // T*B

    const int*   xb    = (const int*)  d_in[0];
    const float* C     = (const float*)d_in[1];
    const float* Uinw  = (const float*)d_in[2];
    const float* Uinb  = (const float*)d_in[3];
    const float* Winw  = (const float*)d_in[4];
    const float* h0in  = (const float*)d_in[5];
    const float* c0in  = (const float*)d_in[6];
    const float* Uhw   = (const float*)d_in[7];
    const float* Uhb   = (const float*)d_in[8];
    const float* Whw   = (const float*)d_in[9];
    const float* Vhw   = (const float*)d_in[10];
    const float* Vhb   = (const float*)d_in[11];
    const float* h0h   = (const float*)d_in[12];
    const float* c0h   = (const float*)d_in[13];
    const float* Why   = (const float*)d_in[14];
    const float* by    = (const float*)d_in[15];
    float* out = (float*)d_out;

    // workspace carve-up
    char* base = (char*)d_ws;
    size_t off = 0;
    auto carve = [&](size_t bytes) {
        size_t o = off; off = (off + bytes + 255) & ~(size_t)255; return o;
    };
    const size_t fragW = 7 * 52 * 512;           // elems per 800-col weight
    bf16*  emb   = (bf16*)(base + carve((size_t)M * 32 * 2));
    bf16*  hall  = (bf16*)(base + carve((size_t)4 * M * 224 * 2));
    float* exbuf = (float*)(base + carve((size_t)3 * M * 208 * 4));
    float* pre   = (float*)(base + carve((size_t)M * 832 * 4));
    bf16*  Usw   = (bf16*)(base + carve((size_t)2 * 52 * 512 * 2));
    bf16*  Wsw   = (bf16*)(base + carve((size_t)4 * fragW * 2));
    bf16*  UhswB = (bf16*)(base + carve((size_t)3 * fragW * 2));
    bf16*  VhswB = (bf16*)(base + carve((size_t)3 * 2 * 13 * 512 * 2));
    bf16*  Ysw   = (bf16*)(base + carve((size_t)4 * 7 * 6 * 512 * 2));
    float* bpre  = (float*)(base + carve((size_t)4 * 832 * 4));
    float* bex   = (float*)(base + carve((size_t)3 * 208 * 4));

    // zero h_bf16_all once per call (covers K-pad columns 208..223)
    hipMemsetAsync(hall, 0, (size_t)4 * M * 224 * 2, stream);

    // embedding gather
    k_embed<<<(M * 32) / 256, 256, 0, stream>>>(xb, C, emb);

    // weight swizzles
    k_swz<<<(2 * 52 * 512) / 256, 256, 0, stream>>>(Uinw, Usw, 24, 800, 52, 1);
    k_swz<<<(int)(fragW / 256), 256, 0, stream>>>(Winw, Wsw, 200, 800, 52, 1);
    for (int n = 0; n < 3; n++) {
        k_swz<<<(int)(fragW / 256), 256, 0, stream>>>(
            Whw + (size_t)n * 200 * 800, Wsw + (size_t)(n + 1) * fragW, 200, 800, 52, 1);
        k_swz<<<(int)(fragW / 256), 256, 0, stream>>>(
            Uhw + (size_t)n * 200 * 800, UhswB + (size_t)n * fragW, 200, 800, 52, 1);
        k_swz<<<(2 * 13 * 512) / 256, 256, 0, stream>>>(
            Vhw + (size_t)n * 24 * 200, VhswB + (size_t)n * 2 * 13 * 512, 24, 200, 13, 0);
    }
    for (int l = 0; l < 4; l++)
        k_swz<<<(7 * 6 * 512) / 256, 256, 0, stream>>>(
            Why + (size_t)l * 200 * 96, Ysw + (size_t)l * 7 * 6 * 512, 200, 96, 6, 0);

    k_bias<<<16, 256, 0, stream>>>(Uinb, Uhb, Vhb, bpre, bex);

    // pre0 = embT @ U_in + b   (2048*52 tiles)
    k_gemm<<<(2048 * 52) / 8, 256, 0, stream>>>(emb, 32, Usw, 52, 2, bpre, pre, 832, 0);
    // ex[n] = tanh(embT @ Vh[n] + b)   (2048*13 tiles each)
    for (int n = 0; n < 3; n++)
        k_gemm<<<(2048 * 13) / 8, 256, 0, stream>>>(
            emb, 32, VhswB + (size_t)n * 2 * 13 * 512, 13, 2, bex + n * 208,
            exbuf + (size_t)n * M * 208, 208, 1);

    // recurrent layers
    for (int n = 0; n < 4; n++) {
        const float* h0 = (n == 0) ? h0in : (h0h + (n - 1) * 200);
        const float* c0 = (n == 0) ? c0in : (c0h + (n - 1) * 200);
        const float* ex = (n == 0) ? (const float*)nullptr
                                   : (exbuf + (size_t)(n - 1) * M * 208);
        k_lstm<<<1, 832, 0, stream>>>(pre, Wsw + (size_t)n * fragW, ex, h0, c0,
                                      hall + (size_t)n * M * 224);
        if (n < 3)
            k_gemm<<<(2048 * 52) / 8, 256, 0, stream>>>(
                hall + (size_t)n * M * 224, 224, UhswB + (size_t)n * fragW, 52, 7,
                bpre + (n + 1) * 832, pre, 832, 0);
    }

    // fused 4-layer output projection   (2048*6 tiles)
    k_logits<<<(2048 * 6) / 8, 256, 0, stream>>>(hall, Ysw, by, out);
}
